// LayoutLMv2RelationExtractionDecoder_58669253264200
// MI455X (gfx1250) — compile-verified, matrix operand order checked by reading, not run on hardware
//
#include <hip/hip_runtime.h>
#include <hip/hip_bf16.h>
#include <math.h>

// ---------------------------------------------------------------------------
// LayoutLMv2 relation-extraction decoder for MI455X (gfx1250, wave32, WMMA)
// ---------------------------------------------------------------------------

typedef __attribute__((ext_vector_type(16))) __bf16 v16bf;
typedef __attribute__((ext_vector_type(8)))  float  v8f;

#define H   768
#define H2  1536
#define HHd 384
#define Bd  32
#define Wd  512
#define Ed  128
#define Rd  1024
#define NROWS (Bd * Rd)          // 32768 relation rows

#define LD_REPR 1560             // 1536 + 24 pad (row stride 780 dw mod 64 = 12)
#define LD_H1   792              // 768  + 24 pad (row stride 396 dw mod 64 = 12)

__device__ __forceinline__ unsigned short f2bf(float f) {
  unsigned int u = __float_as_uint(f);
  unsigned int r = (u + 0x7FFFu + ((u >> 16) & 1u)) >> 16;   // RNE
  return (unsigned short)r;
}
__device__ __forceinline__ float bf2f(unsigned short s) {
  return __uint_as_float(((unsigned int)s) << 16);
}

union FragU { v16bf v; uint4 q[2]; unsigned short s[16]; };

// 16x32 bf16 A/B fragment from a row-major bf16 matrix (global or LDS).
// Lane layout (ISA 7.12.2): lanes 0-15 K in {0..7}u{16..23}, lanes 16-31 +8.
__device__ __forceinline__ v16bf frag_bf16(const unsigned short* base, int ld,
                                           int row_base, int kt, int lane) {
  int r  = row_base + (lane & 15);
  int ko = kt + ((lane >> 4) << 3);
  FragU u;
  u.q[0] = *reinterpret_cast<const uint4*>(base + (size_t)r * ld + ko);
  u.q[1] = *reinterpret_cast<const uint4*>(base + (size_t)r * ld + ko + 16);
  return u.v;
}

// LDS byte offset of a pointer into a __shared__ object (for async-from-LDS).
__device__ __forceinline__ unsigned lds_off_u32(const void* p) {
  return (unsigned)(uintptr_t)(__attribute__((address_space(3))) const void*)p;
}

// ---------------------------------------------------------------------------
__global__ void f32_to_bf16_kernel(const float* __restrict__ src,
                                   unsigned short* __restrict__ dst, int n) {
  int i = blockIdx.x * blockDim.x + threadIdx.x;
  if (i < n) dst[i] = f2bf(src[i]);
}

__global__ void zero_kernel(float* out) {
  if (threadIdx.x == 0 && blockIdx.x == 0) out[0] = 0.0f;
}

// ---------------------------------------------------------------------------
// Fused gather + 2-layer FFNN.  32 rows per block, 256 threads (8 waves).
// Stage1: cooperative gather + f32->bf16 into LDS repr tile [32 x 1536]
// Layer1: [32x1536]x[1536x768] -> ReLU -> LDS bf16 (waves: 2 m-tiles x 4 n-slices)
// Layer2: [32x768]x[768x384]   -> ReLU -> LDS out tile -> async store to global
// ---------------------------------------------------------------------------
__global__ __launch_bounds__(256) void ffnn_kernel(
    const float* __restrict__ hs, const float* __restrict__ eemb,
    const int* __restrict__ rel, const int* __restrict__ est,
    const int* __restrict__ elab,
    const unsigned short* __restrict__ W1b, const float* __restrict__ b1,
    const unsigned short* __restrict__ W2b, const float* __restrict__ b2,
    unsigned short* __restrict__ outw)
{
  __shared__ unsigned short repr[32 * LD_REPR];   // ~100 KB (reused as out tile)
  __shared__ unsigned short h1[32 * LD_H1];       // ~50 KB
  const int tid  = threadIdx.x;
  const int lane = tid & 31;
  const int wave = tid >> 5;
  const int row0 = blockIdx.x * 32;

  // ---- stage 1: cooperative gather + convert (8 threads per row) ----
  {
    int rloc = tid >> 3;
    int tsub = tid & 7;
    int rowg = row0 + rloc;
    int b = rowg >> 10;
    int r = rowg & (Rd - 1);
    int e   = rel[b * Rd + r];
    int tok = est[b * Ed + e];
    int lab = elab[b * Ed + e];
    const float* tokp = hs + (size_t)(b * Wd + tok) * H;
    const float* embp = eemb + (size_t)lab * H;
    unsigned short* dst = repr + rloc * LD_REPR;
#pragma unroll 4
    for (int j = 0; j < 48; ++j) {
      int c = (tsub + j * 8) * 4;                 // 4-float chunk, never crosses 768
      const float* src = (c < H) ? (tokp + c) : (embp + (c - H));
      float4 v = *reinterpret_cast<const float4*>(src);
      ushort4 o;
      o.x = f2bf(v.x); o.y = f2bf(v.y); o.z = f2bf(v.z); o.w = f2bf(v.w);
      *reinterpret_cast<ushort4*>(dst + c) = o;
    }
  }
  __syncthreads();

  const int m = wave >> 2;                        // m-tile: rows m*16 .. m*16+15
  const int s = wave & 3;                         // n-slice

  // ---------------- layer 1 ----------------
  {
    const int nbase = s * 192;                    // 12 n-tiles per wave
    v8f acc[12];
#pragma unroll
    for (int nt = 0; nt < 12; ++nt)
#pragma unroll
      for (int i = 0; i < 8; ++i) acc[nt][i] = 0.0f;

    for (int kt = 0; kt < H2; kt += 32) {
      v16bf a = frag_bf16(repr, LD_REPR, m * 16, kt, lane);  // ds_load_b128 x2
#pragma unroll
      for (int nt = 0; nt < 12; ++nt) {
        v16bf bm = frag_bf16(W1b, H2, nbase + nt * 16, kt, lane);
        acc[nt] = __builtin_amdgcn_wmma_f32_16x16x32_bf16(
            false, a, false, bm, (short)0, acc[nt], false, false);
      }
    }
    int n = lane & 15;
    int mofs = (lane >> 4) * 8;
#pragma unroll
    for (int nt = 0; nt < 12; ++nt) {
      int ncol = nbase + nt * 16 + n;
      float bias = b1[ncol];
#pragma unroll
      for (int i = 0; i < 8; ++i) {
        float v = fmaxf(acc[nt][i] + bias, 0.0f);
        h1[(m * 16 + mofs + i) * LD_H1 + ncol] = f2bf(v);
      }
    }
  }
  __syncthreads();

  // ---------------- layer 2 ----------------
  unsigned short* outt = repr;                    // reuse repr: [32 x 384] packed
  {
    const int nbase = s * 96;                     // 6 n-tiles per wave
    v8f acc2[6];
#pragma unroll
    for (int nt = 0; nt < 6; ++nt)
#pragma unroll
      for (int i = 0; i < 8; ++i) acc2[nt][i] = 0.0f;

    for (int kt = 0; kt < H; kt += 32) {
      v16bf a = frag_bf16(h1, LD_H1, m * 16, kt, lane);
#pragma unroll
      for (int nt = 0; nt < 6; ++nt) {
        v16bf bm = frag_bf16(W2b, H, nbase + nt * 16, kt, lane);
        acc2[nt] = __builtin_amdgcn_wmma_f32_16x16x32_bf16(
            false, a, false, bm, (short)0, acc2[nt], false, false);
      }
    }
    int n = lane & 15;
    int mofs = (lane >> 4) * 8;
#pragma unroll
    for (int nt = 0; nt < 6; ++nt) {
      int ncol = nbase + nt * 16 + n;
      float bias = b2[ncol];
#pragma unroll
      for (int i = 0; i < 8; ++i) {
        float v = fmaxf(acc2[nt][i] + bias, 0.0f);
        outt[(m * 16 + mofs + i) * HHd + ncol] = f2bf(v);
      }
    }
  }
  __syncthreads();

  // ---- async drain: LDS out tile -> global (contiguous 24.5 KB) ----
  {
    const unsigned short* gbase = outw + (size_t)row0 * HHd;
#pragma unroll
    for (int it = 0; it < 6; ++it) {
      int idx = tid + it * 256;                   // 16-byte chunk id (1536 total)
      unsigned long long ga =
          (unsigned long long)(uintptr_t)((const char*)gbase + (size_t)idx * 16);
      unsigned la = lds_off_u32((const char*)outt + (size_t)idx * 16);
      asm volatile("global_store_async_from_lds_b128 %0, %1, off"
                   :: "v"(ga), "v"(la) : "memory");
    }
    asm volatile("s_wait_asynccnt 0x0" ::: "memory");
  }
}

// ---------------------------------------------------------------------------
// Biaffine + CE loss.  32 rows per block; waves split over {o} x {m} x {n-slice}.
// tmp[o][r][i] = sum_j W_bil[o][i][j] * tails[r][j]  (WMMA GEMM)
// logit[o]    = sum_i heads[r][i]*tmp + W_lin terms + b_lin
// ---------------------------------------------------------------------------
__global__ __launch_bounds__(256) void biaffine_loss_kernel(
    const unsigned short* __restrict__ heads,
    const unsigned short* __restrict__ tails,
    const unsigned short* __restrict__ Wbilb,
    const float* __restrict__ Wlin, const float* __restrict__ blin,
    const int* __restrict__ rlabel, float* __restrict__ out)
{
  __shared__ float tmp[2][32][392];               // ~100 KB
  __shared__ float logitsLDS[32][2];
  const int tid  = threadIdx.x;
  const int lane = tid & 31;
  const int wave = tid >> 5;
  const int row0 = blockIdx.x * 32;
  const int o = wave >> 2;                        // output channel
  const int m = (wave >> 1) & 1;                  // m-tile
  const int s = wave & 1;                         // n-slice (2 x 192)
  const int nbase = s * 192;                      // 12 n-tiles
  const unsigned short* Wb = Wbilb + (size_t)o * HHd * HHd;

  v8f acc[12];
#pragma unroll
  for (int nt = 0; nt < 12; ++nt)
#pragma unroll
    for (int i = 0; i < 8; ++i) acc[nt][i] = 0.0f;

  for (int kt = 0; kt < HHd; kt += 32) {
    v16bf a = frag_bf16(tails, HHd, row0 + m * 16, kt, lane);
#pragma unroll
    for (int nt = 0; nt < 12; ++nt) {
      // B column-major == contiguous rows of W_bil[o]
      v16bf bm = frag_bf16(Wb, HHd, nbase + nt * 16, kt, lane);
      acc[nt] = __builtin_amdgcn_wmma_f32_16x16x32_bf16(
          false, a, false, bm, (short)0, acc[nt], false, false);
    }
  }
  {
    int n = lane & 15;
    int mofs = (lane >> 4) * 8;
#pragma unroll
    for (int nt = 0; nt < 12; ++nt)
#pragma unroll
      for (int i = 0; i < 8; ++i)
        tmp[o][m * 16 + mofs + i][nbase + nt * 16 + n] = acc[nt][i];
  }
  __syncthreads();

  // 4 threads per (row, o): dot(heads, tmp) + linear term, shuffle-reduce.
  {
    int g   = tid >> 2;                           // 64 groups
    int row = g >> 1;
    int oo  = g & 1;
    int t   = tid & 3;
    int rowg = row0 + row;
    float sacc = 0.0f;
    for (int i = t; i < HHd; i += 4) {
      float h  = bf2f(heads[(size_t)rowg * HHd + i]);
      float tl = bf2f(tails[(size_t)rowg * HHd + i]);
      sacc += tmp[oo][row][i] * h
            + Wlin[oo * 2 * HHd + i] * h
            + Wlin[oo * 2 * HHd + HHd + i] * tl;
    }
    sacc += __shfl_xor(sacc, 1, 32);
    sacc += __shfl_xor(sacc, 2, 32);
    if (t == 0) logitsLDS[row][oo] = sacc + blin[oo];
  }
  __syncthreads();

  if (tid < 32) {
    int rowg = row0 + tid;
    float l0 = logitsLDS[tid][0];
    float l1 = logitsLDS[tid][1];
    int b = rowg >> 10, r = rowg & (Rd - 1);
    int lbl = rlabel[b * Rd + r];
    float mx = fmaxf(l0, l1);
    float lse = mx + logf(expf(l0 - mx) + expf(l1 - mx));
    float nll = lse - ((lbl == 0) ? l0 : l1);
    atomicAdd(out, nll * (1.0f / (float)Rd));     // sum_b mean_r
  }
}

// ---------------------------------------------------------------------------
extern "C" void kernel_launch(void* const* d_in, const int* in_sizes, int n_in,
                              void* d_out, int out_size, void* d_ws, size_t ws_size,
                              hipStream_t stream)
{
  (void)in_sizes; (void)n_in; (void)out_size; (void)ws_size;
  const float* hs    = (const float*)d_in[0];
  const int*   rhead = (const int*)d_in[1];
  const int*   rtail = (const int*)d_in[2];
  const int*   rlab  = (const int*)d_in[3];
  const int*   est   = (const int*)d_in[4];
  /* d_in[5] entities_end: unused by reference math */
  const int*   elab  = (const int*)d_in[6];
  const float* eemb  = (const float*)d_in[7];
  const float* Wh1   = (const float*)d_in[8];
  const float* bh1   = (const float*)d_in[9];
  const float* Wh2   = (const float*)d_in[10];
  const float* bh2   = (const float*)d_in[11];
  const float* Wt1   = (const float*)d_in[12];
  const float* bt1   = (const float*)d_in[13];
  const float* Wt2   = (const float*)d_in[14];
  const float* bt2   = (const float*)d_in[15];
  const float* Wbil  = (const float*)d_in[16];
  const float* Wlin  = (const float*)d_in[17];
  const float* blin  = (const float*)d_in[18];
  float* out = (float*)d_out;

  unsigned short* ws = (unsigned short*)d_ws;
  size_t off = 0;
  unsigned short* Wh1b = ws + off; off += (size_t)H * H2;
  unsigned short* Wh2b = ws + off; off += (size_t)HHd * H;
  unsigned short* Wt1b = ws + off; off += (size_t)H * H2;
  unsigned short* Wt2b = ws + off; off += (size_t)HHd * H;
  unsigned short* Wbb  = ws + off; off += (size_t)2 * HHd * HHd;
  unsigned short* headsb = ws + off; off += (size_t)NROWS * HHd;
  unsigned short* tailsb = ws + off; off += (size_t)NROWS * HHd;

  // 1) weights -> bf16 (tiny; amortized over 1024 row-tiles)
  f32_to_bf16_kernel<<<(H * H2 + 255) / 256, 256, 0, stream>>>(Wh1, Wh1b, H * H2);
  f32_to_bf16_kernel<<<(HHd * H + 255) / 256, 256, 0, stream>>>(Wh2, Wh2b, HHd * H);
  f32_to_bf16_kernel<<<(H * H2 + 255) / 256, 256, 0, stream>>>(Wt1, Wt1b, H * H2);
  f32_to_bf16_kernel<<<(HHd * H + 255) / 256, 256, 0, stream>>>(Wt2, Wt2b, HHd * H);
  f32_to_bf16_kernel<<<(2 * HHd * HHd + 255) / 256, 256, 0, stream>>>(Wbil, Wbb, 2 * HHd * HHd);
  zero_kernel<<<1, 1, 0, stream>>>(out);

  // 2) fused gather + FFNN for heads and tails (32 rows per block)
  ffnn_kernel<<<NROWS / 32, 256, 0, stream>>>(hs, eemb, rhead, est, elab,
                                              Wh1b, bh1, Wh2b, bh2, headsb);
  ffnn_kernel<<<NROWS / 32, 256, 0, stream>>>(hs, eemb, rtail, est, elab,
                                              Wt1b, bt1, Wt2b, bt2, tailsb);

  // 3) biaffine + CE loss (32 rows per block)
  biaffine_loss_kernel<<<NROWS / 32, 256, 0, stream>>>(headsb, tailsb, Wbb,
                                                       Wlin, blin, rlab, out);
}
